// BitLinearInfer_6330781795073
// MI455X (gfx1250) — matrix operand order repaired and checked by reading, not compiled, not run on hardware
//
#include <hip/hip_runtime.h>
#include <stdint.h>
#include <stddef.h>

// ---- vector types matching the probe-confirmed WMMA builtin signatures ----
typedef _Float16 h4  __attribute__((ext_vector_type(4)));
typedef _Float16 h8  __attribute__((ext_vector_type(8)));
typedef _Float16 h16 __attribute__((ext_vector_type(16)));
typedef float    f4  __attribute__((ext_vector_type(4)));
typedef float    f8  __attribute__((ext_vector_type(8)));
typedef int8_t   b16 __attribute__((ext_vector_type(16)));

#define D_K      4096     // D_IN  (reduction dim)
#define D_N      4096     // D_OUT
#define D_M      8192     // M_TOKENS
#define TILE_M   128      // block tile rows (2 waves x 64)
#define TILE_N   256      // block tile cols (4 waves x 64)
#define KSTEP    32       // one WMMA K per LDS stage
#define LDS_LD   40       // 32 + 8 halves padding (row stride 80B: 16B-aligned,
                          // 20-dword stride -> conflict-free b128 across 64 banks)
#define NT       (D_K / KSTEP)

__global__ __launch_bounds__(256)
void BitLinearInfer_6330781795073_kernel(const float*  __restrict__ x,
                                         const int8_t* __restrict__ wq,
                                         const float*  __restrict__ s,
                                         const float*  __restrict__ bias,
                                         float*        __restrict__ y) {
    // double-buffered f16 tiles: As 2x10KB + Bs 2x20KB = 60KB
    __shared__ _Float16 As[2][TILE_M * LDS_LD];
    __shared__ _Float16 Bs[2][TILE_N * LDS_LD];

    const int tid    = threadIdx.x;
    const int lane   = tid & 31;
    const int wave   = tid >> 5;
    const int waveM  = wave & 1;      // 2 waves along M -> 2*64  = 128
    const int waveN  = wave >> 1;     // 4 waves along N -> 4*64  = 256
    const int blockM = blockIdx.y * TILE_M;
    const int blockN = blockIdx.x * TILE_N;

    const int laneLo = lane & 15;     // position within half-wave
    const int laneHi = lane >> 4;     // 0 = lanes 0-15, 1 = lanes 16-31

    // 64x64 wave tile: 4x4 accumulators (128 VGPRs)
    f8 acc[4][4];
    #pragma unroll
    for (int mi = 0; mi < 4; ++mi)
        #pragma unroll
        for (int ni = 0; ni < 4; ++ni)
            #pragma unroll
            for (int v = 0; v < 8; ++v)
                acc[mi][ni][v] = 0.0f;

    // ---- software pipeline registers (raw global data for next K tile) ----
    f4  xr[4];   // 128 rows x 32 f32  = 1024 f4 chunks / 256 thr = 4 each
    b16 wr[2];   // 256 rows x 32 int8 =  512 b16 chunks / 256 thr = 2 each

    auto loadRegs = [&](int kb0) {
        #pragma unroll
        for (int i = 0; i < 4; ++i) {
            const int idx = i * 256 + tid;
            const int row = idx >> 3;          // 8 float4 per 32-f32 row
            const int cc  = idx & 7;
            xr[i] = *(const f4*)(x + (size_t)(blockM + row) * D_K + kb0 + cc * 4);
        }
        #pragma unroll
        for (int i = 0; i < 2; ++i) {
            const int idx = i * 256 + tid;
            const int row = idx >> 1;          // 2 byte16 per 32-i8 row
            const int cc  = idx & 1;
            wr[i] = *(const b16*)(wq + (size_t)(blockN + row) * D_K + kb0 + cc * 16);
        }
    };

    auto storeLds = [&](int buf) {
        #pragma unroll
        for (int i = 0; i < 4; ++i) {
            const int idx = i * 256 + tid;
            const int row = idx >> 3;
            const int cc  = idx & 7;
            h4 hv;
            #pragma unroll
            for (int j = 0; j < 4; ++j) hv[j] = (_Float16)xr[i][j];
            *(h4*)&As[buf][row * LDS_LD + cc * 4] = hv;
        }
        #pragma unroll
        for (int i = 0; i < 2; ++i) {
            const int idx = i * 256 + tid;
            const int row = idx >> 1;
            const int cc  = idx & 1;
            h8 w0, w1;
            #pragma unroll
            for (int j = 0; j < 8; ++j) {
                w0[j] = (_Float16)(float)wr[i][j];
                w1[j] = (_Float16)(float)wr[i][j + 8];
            }
            _Float16* lp = &Bs[buf][row * LDS_LD + cc * 16];
            *(h8*)(lp)     = w0;
            *(h8*)(lp + 8) = w1;
        }
    };

    auto computeTile = [&](int buf) {
        h16 afr[4];
        h16 bfr[4];
        // A fragment: lane = row m (laneLo); chunks K=[kc..kc+7],[kc+16..kc+23]
        #pragma unroll
        for (int mi = 0; mi < 4; ++mi) {
            const int row = waveM * 64 + mi * 16 + laneLo;
            const int kc  = laneHi << 3;               // +8 for hi lanes
            const _Float16* ap = &As[buf][row * LDS_LD + kc];
            h8 lo = *(const h8*)(ap);
            h8 hi = *(const h8*)(ap + 16);
            afr[mi] = __builtin_shufflevector(lo, hi,
                0,1,2,3,4,5,6,7,8,9,10,11,12,13,14,15);
        }
        // B fragment: lane = col n (laneLo); chunks K=[kc..kc+7],[kc+16..kc+23]
        #pragma unroll
        for (int ni = 0; ni < 4; ++ni) {
            const int n  = waveN * 64 + ni * 16 + laneLo;
            const int kc = laneHi << 3;                // +8 for hi lanes
            const _Float16* bp = &Bs[buf][n * LDS_LD + kc];
            h8 b0 = *(const h8*)(bp);
            h8 b1 = *(const h8*)(bp + 16);
            bfr[ni] = __builtin_shufflevector(b0, b1,
                0,1,2,3,4,5,6,7,8,9,10,11,12,13,14,15);
        }
        #pragma unroll
        for (int mi = 0; mi < 4; ++mi)
            #pragma unroll
            for (int ni = 0; ni < 4; ++ni)
                acc[mi][ni] = __builtin_amdgcn_wmma_f32_16x16x32_f16(
                    /*neg_a=*/false, afr[mi],
                    /*neg_b=*/false, bfr[ni],
                    /*c_mod=*/(short)0, acc[mi][ni],
                    /*reuse_a=*/false, /*reuse_b=*/false);
    };

    // ---------------- pipelined main loop: one barrier per K-step ----------
    loadRegs(0);
    storeLds(0);
    __syncthreads();

    #pragma unroll 1
    for (int t = 0; t < NT; ++t) {
        const int kb_next = (t + 1) * KSTEP;
        if (kb_next < D_K) loadRegs(kb_next);   // global loads overlap compute
        computeTile(t & 1);                      // ds_load_b128 + 16x wmma
        if (kb_next < D_K) storeLds((t + 1) & 1);
        __syncthreads();                         // writes to buf t+1 never race
                                                 // reads of buf t within an iter
    }

    // ---------------- epilogue: y = acc * s + bias -------------------------
    // C layout: VGPR v -> M = v (lanes 0-15) / v+8 (lanes 16-31), N = lane%16
    const float sv = s[0];
    #pragma unroll
    for (int mi = 0; mi < 4; ++mi) {
        const int row0 = blockM + waveM * 64 + mi * 16 + (laneHi << 3);
        #pragma unroll
        for (int ni = 0; ni < 4; ++ni) {
            const int col = blockN + waveN * 64 + ni * 16 + laneLo;
            const float bc = bias[col];
            f8 c = acc[mi][ni];
            #pragma unroll
            for (int v = 0; v < 8; ++v)
                y[(size_t)(row0 + v) * D_N + col] = c[v] * sv + bc;
        }
    }
}

extern "C" void kernel_launch(void* const* d_in, const int* in_sizes, int n_in,
                              void* d_out, int out_size, void* d_ws, size_t ws_size,
                              hipStream_t stream) {
    (void)in_sizes; (void)n_in; (void)out_size; (void)d_ws; (void)ws_size;
    const float*  x    = (const float*)d_in[0];
    const int8_t* wq   = (const int8_t*)d_in[1];
    const float*  sc   = (const float*)d_in[2];
    const float*  bias = (const float*)d_in[3];
    float*        y    = (float*)d_out;

    dim3 grid(D_N / TILE_N, D_M / TILE_M);   // (16, 64)
    dim3 block(256, 1, 1);                   // 8 wave32s
    BitLinearInfer_6330781795073_kernel<<<grid, block, 0, stream>>>(x, wq, sc, bias, y);
}
